// GNNLayer_35880156791229
// MI455X (gfx1250) — matrix operand order repaired and checked by reading, not compiled, not run on hardware
//
#include <hip/hip_runtime.h>
#include <hip/hip_bf16.h>
#include <math.h>

#define NN    50000
#define EE    600000
#define DD    128
#define HH    8
#define DHH   16
#define FFDIM 512

typedef __attribute__((ext_vector_type(16))) __bf16 v16bf;
typedef __attribute__((ext_vector_type(8)))  float  v8f;

// ---------------------------------------------------------------------------
// Weight pre-conversion: f32 [K,N] row-major  ->  bf16 fragment-tiled layout.
// Tile index kt = k/16. Element (k,n) stored at Wt[(kt*N + n)*16 + (k%16)],
// so a B-fragment lane (column n, K-run of 16) is one contiguous 32B load.
// ---------------------------------------------------------------------------
__global__ void __launch_bounds__(256) convert_weight_tiled(
    const float* __restrict__ W, __bf16* __restrict__ Wt, int K, int N)
{
    int idx = blockIdx.x * blockDim.x + threadIdx.x;
    if (idx >= K * N) return;
    int k = idx / N;
    int n = idx - k * N;
    Wt[((size_t)(k >> 4) * N + n) * 16 + (k & 15)] = (__bf16)W[idx];
}

// ---------------------------------------------------------------------------
// WMMA GEMM:  C[M,N] = A[M,K] @ W[K,N] (+bias) (+relu)
// A: f32 (converted to bf16 in registers), W: pre-tiled bf16, C: f32.
// One wave computes a 16(M) x 128(N) strip: 8 accumulators, K steps of 32.
// All 8 B-fragments of a k-step are loaded up-front (one 16x b128 clause) so
// the 8 WMMAs overlap the in-flight loads instead of waiting one-by-one.
// A-fragment (16-bit A 16x32, contiguous k order, same as B): lane L holds row M=L%16;
//   lanes 0-15:  K = k0+0..15
//   lanes 16-31: K = k0+16..31   -> two contiguous float4 pairs.
// B-fragment: lane L holds column n, K = k0+(L/16)*16+i  -> one v16bf load.
// D: element v of lane L -> row = row0 + (L/16)*8 + v, col = n.
// ---------------------------------------------------------------------------
__global__ void __launch_bounds__(256) gemm_wmma_bf16(
    const float* __restrict__ A, const __bf16* __restrict__ Wt,
    const float* __restrict__ bias, float* __restrict__ C,
    int M, int N, int K, int doRelu)
{
    const int wavesPerBlock = blockDim.x >> 5;
    int wave = blockIdx.x * wavesPerBlock + (threadIdx.x >> 5);
    int lane = threadIdx.x & 31;
    int colStrips = N >> 7;                 // 128-wide strips
    int tiles = (M >> 4) * colStrips;
    if (wave >= tiles) return;
    int rowTile  = wave / colStrips;
    int colStrip = wave - rowTile * colStrips;
    int row0 = rowTile << 4;
    int n0   = colStrip << 7;
    int g  = lane >> 4;   // half-wave id
    int lr = lane & 15;   // row (A/D) or column (B/D) within 16-tile

    v8f zero = {0.f, 0.f, 0.f, 0.f, 0.f, 0.f, 0.f, 0.f};
    v8f acc[8];
    #pragma unroll
    for (int c = 0; c < 8; ++c) acc[c] = zero;

    const float* Arow = A + (size_t)(row0 + lr) * K;
    const v16bf* Wv16 = (const v16bf*)Wt;

    for (int k0 = 0; k0 < K; k0 += 32) {
        if (k0 + 32 < K)
            __builtin_prefetch(Arow + k0 + 64, 0, 1);   // global_prefetch_b8

        // ---- issue all 8 B-fragment loads first (one long clause) ----
        size_t ktBase = (size_t)((k0 >> 4) + g) * N;
        v16bf bf[8];
        #pragma unroll
        for (int c = 0; c < 8; ++c)
            bf[c] = Wv16[ktBase + (n0 + (c << 4) + lr)];

        // ---- A fragment (16x32 bf16) from two contiguous float4 pairs ----
        v16bf afrag;
        const float* p0 = Arow + k0 + (g << 4);
        const float* p1 = p0 + 8;
        float4 a0 = *(const float4*)(p0);
        float4 a1 = *(const float4*)(p0 + 4);
        float4 b0 = *(const float4*)(p1);
        float4 b1 = *(const float4*)(p1 + 4);
        afrag[0]  = (__bf16)a0.x; afrag[1]  = (__bf16)a0.y;
        afrag[2]  = (__bf16)a0.z; afrag[3]  = (__bf16)a0.w;
        afrag[4]  = (__bf16)a1.x; afrag[5]  = (__bf16)a1.y;
        afrag[6]  = (__bf16)a1.z; afrag[7]  = (__bf16)a1.w;
        afrag[8]  = (__bf16)b0.x; afrag[9]  = (__bf16)b0.y;
        afrag[10] = (__bf16)b0.z; afrag[11] = (__bf16)b0.w;
        afrag[12] = (__bf16)b1.x; afrag[13] = (__bf16)b1.y;
        afrag[14] = (__bf16)b1.z; afrag[15] = (__bf16)b1.w;

        // ---- 8 back-to-back WMMAs, overlapping remaining B loads ----
        #pragma unroll
        for (int c = 0; c < 8; ++c)
            acc[c] = __builtin_amdgcn_wmma_f32_16x16x32_bf16(
                false, afrag, false, bf[c], (short)0, acc[c], false, false);
    }

    // ---- epilogue: bias, relu, store ----
    #pragma unroll
    for (int c = 0; c < 8; ++c) {
        int n = n0 + (c << 4) + lr;
        float b = bias ? bias[n] : 0.0f;
        #pragma unroll
        for (int v = 0; v < 8; ++v) {
            float val = acc[c][v] + b;
            if (doRelu) val = fmaxf(val, 0.0f);
            C[(size_t)(row0 + (g << 3) + v) * N + n] = val;
        }
    }
}

// ---------------------------------------------------------------------------
// Attention helpers: monotone f32<->u32 mapping for atomic float max
// ---------------------------------------------------------------------------
__device__ __forceinline__ unsigned mapf(float f) {
    unsigned u = __float_as_uint(f);
    return (u & 0x80000000u) ? ~u : (u | 0x80000000u);
}
__device__ __forceinline__ float unmapf(unsigned u) {
    return (u & 0x80000000u) ? __uint_as_float(u & 0x7FFFFFFFu)
                             : __uint_as_float(~u);
}

__global__ void __launch_bounds__(256) init_attn(float* Ob, unsigned* mB, float* den) {
    size_t i = (size_t)blockIdx.x * blockDim.x + threadIdx.x;
    if (i < (size_t)NN * DD) Ob[i] = 0.0f;
    if (i < (size_t)NN * HH) { mB[i] = 0u; den[i] = 0.0f; }
}

// logit = dot(K[src], Q[dst] + e) / sqrt(DH); store logit; atomic segment max
__global__ void __launch_bounds__(256) edge_logits(
    const float* __restrict__ Kb, const float* __restrict__ Qb,
    const float* __restrict__ Eb, const int* __restrict__ srcI,
    const int* __restrict__ dstI, float* __restrict__ alphaB,
    unsigned* __restrict__ mB)
{
    int id = blockIdx.x * blockDim.x + threadIdx.x;
    if (id >= EE * HH) return;
    int h = id & (HH - 1);
    int e = id >> 3;
    int s = srcI[e], d = dstI[e];
    const float4* kp = (const float4*)(Kb + (size_t)s * DD + h * DHH);
    const float4* qp = (const float4*)(Qb + (size_t)d * DD + h * DHH);
    const float4* ep = (const float4*)(Eb + (size_t)e * DD + h * DHH);
    float acc = 0.0f;
    #pragma unroll
    for (int i = 0; i < 4; ++i) {
        float4 k = kp[i], q = qp[i], ee = ep[i];
        acc += k.x * (q.x + ee.x) + k.y * (q.y + ee.y)
             + k.z * (q.z + ee.z) + k.w * (q.w + ee.w);
    }
    float l = acc * 0.25f;           // 1/sqrt(16)
    alphaB[id] = l;
    atomicMax(mB + (size_t)d * HH + h, mapf(l));
}

// alpha = exp(logit - m[dst]); atomic segment sum
__global__ void __launch_bounds__(256) edge_expsum(
    const int* __restrict__ dstI, const unsigned* __restrict__ mB,
    float* __restrict__ den, float* __restrict__ alphaB)
{
    int id = blockIdx.x * blockDim.x + threadIdx.x;
    if (id >= EE * HH) return;
    int h = id & (HH - 1);
    int e = id >> 3;
    int d = dstI[e];
    float m = unmapf(mB[(size_t)d * HH + h]);
    float a = __expf(alphaB[id] - m);
    alphaB[id] = a;
    atomicAdd(den + (size_t)d * HH + h, a);
}

// alpha /= den[dst]; message scatter: out[dst] += (V[src]+e) * alpha
__global__ void __launch_bounds__(256) edge_scatter(
    const float* __restrict__ Vb, const float* __restrict__ Eb,
    const float* __restrict__ den, const int* __restrict__ srcI,
    const int* __restrict__ dstI, float* __restrict__ alphaB,
    float* __restrict__ Ob)
{
    int id = blockIdx.x * blockDim.x + threadIdx.x;
    if (id >= EE * HH) return;
    int h = id & (HH - 1);
    int e = id >> 3;
    int s = srcI[e], d = dstI[e];
    float a = alphaB[id] / (den[(size_t)d * HH + h] + 1e-16f);
    alphaB[id] = a;                   // final alpha output (in d_out)
    const float* vp = Vb + (size_t)s * DD + h * DHH;
    const float* ep = Eb + (size_t)e * DD + h * DHH;
    float* op = Ob + (size_t)d * DD + h * DHH;
    #pragma unroll
    for (int j = 0; j < DHH; ++j)
        atomicAdd(op + j, (vp[j] + ep[j]) * a);   // global_atomic_add_f32
}

__global__ void __launch_bounds__(256) add_skip(float* __restrict__ Ob,
                                                const float* __restrict__ Sb) {
    size_t i = (size_t)blockIdx.x * blockDim.x + threadIdx.x;
    if (i < (size_t)NN * DD) Ob[i] += Sb[i];
}

// ---------------------------------------------------------------------------
// BatchNorm (training-mode batch stats over node dim, biased var), C = 128
// ---------------------------------------------------------------------------
__global__ void __launch_bounds__(128) zero_stats(float* accS, float* accQ) {
    int i = threadIdx.x;
    if (i < DD) { accS[i] = 0.0f; accQ[i] = 0.0f; }
}

__global__ void __launch_bounds__(256) bn_stats(const float* __restrict__ X,
                                                float* accS, float* accQ, int M) {
    __shared__ float ls[DD], lq[DD];
    for (int i = threadIdx.x; i < DD; i += blockDim.x) { ls[i] = 0.f; lq[i] = 0.f; }
    __syncthreads();
    size_t total = (size_t)M * DD;
    size_t stride = (size_t)gridDim.x * blockDim.x;
    for (size_t i = (size_t)blockIdx.x * blockDim.x + threadIdx.x; i < total; i += stride) {
        float v = X[i];
        int c = (int)(i & (DD - 1));
        atomicAdd(&ls[c], v);         // ds_add_f32, conflict-free per wave
        atomicAdd(&lq[c], v * v);
    }
    __syncthreads();
    for (int i = threadIdx.x; i < DD; i += blockDim.x) {
        atomicAdd(&accS[i], ls[i]);
        atomicAdd(&accQ[i], lq[i]);
    }
}

__global__ void __launch_bounds__(128) bn_finalize(const float* accS, const float* accQ,
                                                   float* mean, float* var, int M) {
    int i = threadIdx.x;
    if (i < DD) {
        float mu = accS[i] / (float)M;
        mean[i] = mu;
        var[i]  = accQ[i] / (float)M - mu * mu;
    }
}

__global__ void __launch_bounds__(256) bn_apply(float* __restrict__ X,
                                                const float* __restrict__ mean,
                                                const float* __restrict__ var,
                                                const float* __restrict__ gamma,
                                                const float* __restrict__ beta, int M) {
    size_t i = (size_t)blockIdx.x * blockDim.x + threadIdx.x;
    if (i >= (size_t)M * DD) return;
    int c = (int)(i & (DD - 1));
    X[i] = (X[i] - mean[c]) * rsqrtf(var[c] + 1e-5f) * gamma[c] + beta[c];
}

// ---------------------------------------------------------------------------
extern "C" void kernel_launch(void* const* d_in, const int* in_sizes, int n_in,
                              void* d_out, int out_size, void* d_ws, size_t ws_size,
                              hipStream_t stream) {
    const float* x   = (const float*)d_in[0];
    const int*  eidx = (const int*)  d_in[1];
    const float* ea  = (const float*)d_in[2];
    const float* Wk  = (const float*)d_in[3];
    const float* bk  = (const float*)d_in[4];
    const float* Wq  = (const float*)d_in[5];
    const float* bq  = (const float*)d_in[6];
    const float* Wv  = (const float*)d_in[7];
    const float* bv  = (const float*)d_in[8];
    const float* We  = (const float*)d_in[9];
    const float* Wsk = (const float*)d_in[10];
    const float* bsk = (const float*)d_in[11];
    const float* g1  = (const float*)d_in[12];
    const float* be1 = (const float*)d_in[13];
    const float* W1  = (const float*)d_in[14];
    const float* bf1 = (const float*)d_in[15];
    const float* W2  = (const float*)d_in[16];
    const float* bf2 = (const float*)d_in[17];
    const float* g2  = (const float*)d_in[18];
    const float* be2 = (const float*)d_in[19];
    (void)in_sizes; (void)n_in; (void)out_size; (void)ws_size;

    float* y     = (float*)d_out;                 // [N, D]
    float* alpha = y + (size_t)NN * DD;           // [E, H]

    // workspace carve-out
    float* Kb = (float*)d_ws;                     // N*128
    float* Qb = Kb + (size_t)NN * DD;             // N*128
    float* Vb = Qb + (size_t)NN * DD;             // N*128
    float* Sb = Vb + (size_t)NN * DD;             // N*128 (skip)
    float* Hb = Kb;                               // FFN hidden reuses K/Q/V/S = N*512
    float* Eb = Sb + (size_t)NN * DD;             // E*128
    float* Ob = Eb + (size_t)EE * DD;             // N*128 (attention out)
    unsigned* mB = (unsigned*)(Ob + (size_t)NN * DD); // N*H  mapped max
    float* den  = (float*)(mB + (size_t)NN * HH);     // N*H
    float* accS = den + (size_t)NN * HH;
    float* accQ = accS + DD;
    float* mu   = accQ + DD;
    float* var  = mu + DD;

    // tiled bf16 weights (32B-aligned for v16bf loads)
    uintptr_t wtb = ((uintptr_t)(var + DD) + 31) & ~(uintptr_t)31;
    __bf16* WkT  = (__bf16*)wtb;                  // 128*128
    __bf16* WqT  = WkT  + (size_t)DD * DD;
    __bf16* WvT  = WqT  + (size_t)DD * DD;
    __bf16* WskT = WvT  + (size_t)DD * DD;
    __bf16* WeT  = WskT + (size_t)DD * DD;
    __bf16* W1T  = WeT  + (size_t)DD * DD;        // 128*512
    __bf16* W2T  = W1T  + (size_t)DD * FFDIM;     // 512*128

    const int* srcI = eidx;
    const int* dstI = eidx + EE;

    dim3 B(256);
    auto gemmBlocks = [](int M, int Ncol) {
        int tiles = (M >> 4) * (Ncol >> 7);       // 16x128 strip per wave
        return (tiles + 7) / 8;                   // 8 waves per 256-thread block
    };
    int nBlocksND = (NN * DD + 255) / 256;
    int ehBlocks  = (EE * HH + 255) / 256;
    int cwB  = (DD * DD + 255) / 256;
    int cwB1 = (DD * FFDIM + 255) / 256;

    // weight pre-conversion to fragment-tiled bf16 (one-time, tiny)
    convert_weight_tiled<<<cwB,  B, 0, stream>>>(Wk,  WkT,  DD, DD);
    convert_weight_tiled<<<cwB,  B, 0, stream>>>(Wq,  WqT,  DD, DD);
    convert_weight_tiled<<<cwB,  B, 0, stream>>>(Wv,  WvT,  DD, DD);
    convert_weight_tiled<<<cwB,  B, 0, stream>>>(Wsk, WskT, DD, DD);
    convert_weight_tiled<<<cwB,  B, 0, stream>>>(We,  WeT,  DD, DD);
    convert_weight_tiled<<<cwB1, B, 0, stream>>>(W1,  W1T,  DD, FFDIM);
    convert_weight_tiled<<<cwB1, B, 0, stream>>>(W2,  W2T,  FFDIM, DD);

    // node / edge projections (bf16 WMMA GEMMs)
    gemm_wmma_bf16<<<gemmBlocks(NN, DD), B, 0, stream>>>(x,  WkT,  bk,  Kb, NN, DD, DD, 0);
    gemm_wmma_bf16<<<gemmBlocks(NN, DD), B, 0, stream>>>(x,  WqT,  bq,  Qb, NN, DD, DD, 0);
    gemm_wmma_bf16<<<gemmBlocks(NN, DD), B, 0, stream>>>(x,  WvT,  bv,  Vb, NN, DD, DD, 0);
    gemm_wmma_bf16<<<gemmBlocks(NN, DD), B, 0, stream>>>(x,  WskT, bsk, Sb, NN, DD, DD, 0);
    gemm_wmma_bf16<<<gemmBlocks(EE, DD), B, 0, stream>>>(ea, WeT, nullptr, Eb, EE, DD, DD, 0);

    // segment softmax attention + scatter
    init_attn<<<nBlocksND, B, 0, stream>>>(Ob, mB, den);
    edge_logits <<<ehBlocks, B, 0, stream>>>(Kb, Qb, Eb, srcI, dstI, alpha, mB);
    edge_expsum <<<ehBlocks, B, 0, stream>>>(dstI, mB, den, alpha);
    edge_scatter<<<ehBlocks, B, 0, stream>>>(Vb, Eb, den, srcI, dstI, alpha, Ob);

    // skip + BN1
    add_skip<<<nBlocksND, B, 0, stream>>>(Ob, Sb);
    zero_stats <<<1, 128, 0, stream>>>(accS, accQ);
    bn_stats   <<<512, B, 0, stream>>>(Ob, accS, accQ, NN);
    bn_finalize<<<1, 128, 0, stream>>>(accS, accQ, mu, var, NN);
    bn_apply   <<<nBlocksND, B, 0, stream>>>(Ob, mu, var, g1, be1, NN);

    // FFN: relu(Ob@W1+bf1) @ W2 + bf2, then BN2 (in place in d_out)
    gemm_wmma_bf16<<<gemmBlocks(NN, FFDIM), B, 0, stream>>>(Ob, W1T, bf1, Hb, NN, FFDIM, DD, 1);
    gemm_wmma_bf16<<<gemmBlocks(NN, DD),    B, 0, stream>>>(Hb, W2T, bf2, y,  NN, DD, FFDIM, 0);

    zero_stats <<<1, 128, 0, stream>>>(accS, accQ);
    bn_stats   <<<512, B, 0, stream>>>(y, accS, accQ, NN);
    bn_finalize<<<1, 128, 0, stream>>>(accS, accQ, mu, var, NN);
    bn_apply   <<<nBlocksND, B, 0, stream>>>(y, mu, var, g2, be2, NN);
}
